// MultiHeadAttentionLayer_90374701842915
// MI455X (gfx1250) — matrix operand order repaired and checked by reading, not compiled, not run on hardware
//
#include <hip/hip_runtime.h>

// ---------------------------------------------------------------------------
// Problem constants (match reference): B=2, N=512, IN_DIM=64, H=8, D=8, HD=64
// ---------------------------------------------------------------------------
#define BB   2
#define NN   512
#define HD   64
#define RTOT (BB * NN)   // 1024 rows for the fused projection GEMM
#define IT   2           // i-rows per fused block

typedef __attribute__((ext_vector_type(16))) _Float16 v16h;
typedef __attribute__((ext_vector_type(8)))  float    v8f;

// ---------------------------------------------------------------------------
// Kernel 1: QKV projection via split-f16 WMMA (f32-accurate).
//   Out[r, c] = sum_k h[r, k] * W[c, k],  r in [0,1024), c,k in [0,64)
//   Each f32 operand is decomposed x = hi + lo (f16 each); product accumulated
//   as A_lo*B_hi + A_hi*B_lo + A_hi*B_hi in the f32 WMMA accumulator, so only
//   the ~2^-22 lo*lo term is dropped.
//   K output additionally scaled by D**-0.5 (folded here, f32, post-WMMA).
// One wave computes one 16x16 tile: 2 k-steps x 3 WMMAs = 6 v_wmma.
// grid = (RTOT/16, 64/16, 3) = (64, 4, 3), block = 32.
// ---------------------------------------------------------------------------
__global__ __launch_bounds__(32) void qkv_proj_kernel(
    const float* __restrict__ hin,
    const float* __restrict__ WQ,
    const float* __restrict__ WK,
    const float* __restrict__ WV,
    float* __restrict__ qkv)        // [3][RTOT][64] in workspace
{
    const int lane = threadIdx.x;            // 0..31
    const int m0   = blockIdx.x * 16;        // row tile
    const int c0   = blockIdx.y * 16;        // col tile
    const int mat  = blockIdx.z;             // 0=Q, 1=K, 2=V
    const float* W = (mat == 0) ? WQ : (mat == 1) ? WK : WV;

    const int lm   = lane & 15;
    const int half = lane >> 4;

    const float* arow = hin + (size_t)(m0 + lm) * 64;

    v8f acc = {};
    #pragma unroll
    for (int k0 = 0; k0 < 64; k0 += 32) {
        v16h a_hi, a_lo, b_hi, b_lo;
        // A (16x32 f16): lane lm = M; lanes 16-31 start at K=8 within the
        // step; element e -> K = k0 + 8*half + 16*(e>>3) + (e&7) (ISA 7.12.2)
        #pragma unroll
        for (int e = 0; e < 16; ++e) {
            const int kk = k0 + half * 8 + ((e >> 3) << 4) + (e & 7);
            const float av = arow[kk];
            const _Float16 ah = (_Float16)av;
            a_hi[e] = ah;
            a_lo[e] = (_Float16)(av - (float)ah);
        }
        // B (32x16 f16): lane lm = N (output channel); element e -> same K as A.
        // B[k][n] = W[c0+n][k].
        const float* brow = W + (size_t)(c0 + lm) * 64;
        #pragma unroll
        for (int e = 0; e < 16; ++e) {
            const float bv = brow[k0 + half * 8 + ((e >> 3) << 4) + (e & 7)];
            const _Float16 bh = (_Float16)bv;
            b_hi[e] = bh;
            b_lo[e] = (_Float16)(bv - (float)bh);
        }
        // 8 args: (neg_a, A, neg_b, B, c_mod, C, reuse_a, reuse_b)
        acc = __builtin_amdgcn_wmma_f32_16x16x32_f16(
            false, a_lo, false, b_hi, (short)0, acc, false, false);
        acc = __builtin_amdgcn_wmma_f32_16x16x32_f16(
            false, a_hi, false, b_lo, (short)0, acc, false, false);
        acc = __builtin_amdgcn_wmma_f32_16x16x32_f16(
            false, a_hi, false, b_hi, (short)0, acc, false, false);
    }

    const float scale = (mat == 1) ? 0.35355339059327373f : 1.0f; // D**-0.5
    float* outp = qkv + (size_t)mat * RTOT * 64;
    // D layout: VGPR v -> row m0 + v + 8*half, lane lm -> col c0 + lm
    #pragma unroll
    for (int v = 0; v < 8; ++v) {
        const int row = m0 + v + 8 * half;
        outp[(size_t)row * 64 + (c0 + lm)] = acc[v] * scale;
    }
}

// ---------------------------------------------------------------------------
// Kernel 2: fused scores + online softmax + aggregation (HBM-bound stream).
// Block = 128 threads handles (b, two rows i, all 64 channels).
//   phase A: preload Q rows into LDS
//   phase B: scores s[h][il][j] = Q(i,h,:).K(j,h,:) + (mask? 0 : -1e24) in LDS
//   phase C: per (i, c=h*8+d) online softmax over j, streaming e_att/e_value
//            exactly once with non-temporal loads (one v_exp per element);
//            out = sum_j p_j*mask_j*(V_jc + Ev_cj).  The post-softmax mask is
//            recovered from the score (< -1e20 <=> masked).
// grid = (NN/IT, BB) = (256, 2), block = 128.
// ---------------------------------------------------------------------------
__global__ __launch_bounds__(128) void fused_attn_kernel(
    const float* __restrict__ qkv,      // [3][RTOT][64]
    const float* __restrict__ e_att,    // [B][N][N][64]
    const float* __restrict__ e_value,  // [B][N][N][64]
    const int*   __restrict__ amask,    // [B][N][N] (bool as int32)
    float* __restrict__ out)            // [B][N][64]
{
    __shared__ float s_sc[8][IT][513];  // padded: conflict-free read & write
    __shared__ float s_q[IT * 64];

    const int b  = blockIdx.y;
    const int i0 = blockIdx.x * IT;
    const int t  = threadIdx.x;

    const float* Q = qkv;
    const float* K = qkv + (size_t)1 * RTOT * 64;
    const float* V = qkv + (size_t)2 * RTOT * 64;

    // ---- phase A: Q tile (IT*64 = 128 floats) ----
    s_q[t] = Q[(size_t)(b * NN + i0 + (t >> 6)) * 64 + (t & 63)];
    __syncthreads();

    // ---- phase B: scores into LDS (K pre-scaled by D**-0.5) ----
    for (int f = t; f < 8 * IT * NN; f += 128) {
        const int j  = f & 511;
        const int il = (f >> 9) & (IT - 1);
        const int hh = f >> 10;
        const float* kr = &K[(size_t)(b * NN + j) * 64 + hh * 8];
        const float* qr = &s_q[il * 64 + hh * 8];
        float dot = 0.f;
        #pragma unroll
        for (int d = 0; d < 8; ++d) dot += qr[d] * kr[d];
        const int mv = amask[(size_t)(b * NN + i0 + il) * NN + j];
        s_sc[hh][il][j] = dot + (mv ? 0.f : -1e24f);
    }
    __syncthreads();

    // ---- phase C: online softmax + aggregation, single pass over j ----
    const int c  = t & 63;          // channel = h*8 + d
    const int il = t >> 6;
    const int ig = i0 + il;
    const int hh = c >> 3;
    const size_t ebase = ((size_t)(b * NN + ig) * NN) * 64 + c;
    const float* Vb = &V[(size_t)(b * NN) * 64 + c];

    float m = -INFINITY, s = 0.f, acc = 0.f;
    #pragma unroll 4
    for (int j = 0; j < NN; ++j) {
        const float ea = __builtin_nontemporal_load(e_att   + ebase + (size_t)j * 64);
        const float ev = __builtin_nontemporal_load(e_value + ebase + (size_t)j * 64);
        const float sj = s_sc[hh][il][j];
        const float vj = Vb[(size_t)j * 64];
        const float mk = (sj < -1e20f) ? 0.f : 1.f;   // post-softmax mask
        const float x  = sj + ea;
        // one-exp online softmax:
        //   d>0: new max -> rescale old state by w; new term weight 1
        //   d<=0: keep max  -> new term weight w
        const float d  = x - m;
        const float w  = __expf(-fabsf(d));
        const bool  gt = d > 0.f;
        const float e  = gt ? 1.f : w;
        const float sc = gt ? w : 1.f;
        if (gt) m = x;
        s   = s * sc + e;
        acc = acc * sc + e * mk * (vj + ev);
    }
    __builtin_nontemporal_store(acc / s, out + (size_t)(b * NN + ig) * 64 + c);
}

// ---------------------------------------------------------------------------
// Launch
// ---------------------------------------------------------------------------
extern "C" void kernel_launch(void* const* d_in, const int* in_sizes, int n_in,
                              void* d_out, int out_size, void* d_ws, size_t ws_size,
                              hipStream_t stream)
{
    (void)in_sizes; (void)n_in; (void)out_size; (void)ws_size;

    const float* h       = (const float*)d_in[0];
    const float* e_att   = (const float*)d_in[1];
    const float* e_value = (const float*)d_in[2];
    const int*   amask   = (const int*)d_in[3];
    const float* WQ      = (const float*)d_in[4];
    const float* WK      = (const float*)d_in[5];
    const float* WV      = (const float*)d_in[6];
    float*       outp    = (float*)d_out;
    float*       qkv     = (float*)d_ws;   // 3 * 1024 * 64 floats = 768 KB

    dim3 g1(RTOT / 16, HD / 16, 3);        // (64, 4, 3), 1 wave per tile
    qkv_proj_kernel<<<g1, 32, 0, stream>>>(h, WQ, WK, WV, qkv);

    dim3 g2(NN / IT, BB);                  // (256, 2)
    fused_attn_kernel<<<g2, 128, 0, stream>>>(qkv, e_att, e_value, amask, outp);
}
